// TRIZRL_29592324670159
// MI455X (gfx1250) — matrix operand order repaired and checked by reading, not compiled, not run on hardware
//
#include <hip/hip_runtime.h>
#include <hip/hip_bf16.h>

typedef __attribute__((ext_vector_type(16))) _Float16 v16h;
typedef __attribute__((ext_vector_type(8)))  _Float16 v8h;
typedef __attribute__((ext_vector_type(8)))  float    v8f;

#define BATCH 131072
#define INP 5
#define HID 64
#define NEXP 40
#define OUTD 5

// ---------------------------------------------------------------------------
// Kernel 1: encoder (tanh(x@enc_w+b)) -> zh (f16, WMMA feed) + policy logits
// One block = 64 rows, 256 threads.
// ---------------------------------------------------------------------------
__global__ void __launch_bounds__(256) enc_logits_kernel(
    const float* __restrict__ x, const float* __restrict__ enc_w,
    const float* __restrict__ enc_b, const float* __restrict__ pol_w,
    const float* __restrict__ pol_b, _Float16* __restrict__ zh,
    float* __restrict__ logits)
{
    __shared__ float zs[64][HID + 1];  // +1 pad: avoid bank conflicts on column reads
    const int tid = threadIdx.x;
    const long row0 = (long)blockIdx.x * 64;

    // Phase 1: 64 rows x 64 hid = 4096 elems / 256 threads = 16 each
    #pragma unroll
    for (int i = 0; i < 16; ++i) {
        int idx = i * 256 + tid;
        int r = idx >> 6, c = idx & 63;
        long b = row0 + r;
        const float* xp = x + b * INP;
        float acc = enc_b[c];
        #pragma unroll
        for (int d = 0; d < INP; ++d) acc += xp[d] * enc_w[d * HID + c];
        float zv = tanhf(acc);
        zs[r][c] = zv;
        zh[b * HID + c] = (_Float16)zv;
    }
    __syncthreads();

    // Phase 2: 64 rows x 40 logits = 2560 / 256 = 10 each
    #pragma unroll
    for (int i = 0; i < 10; ++i) {
        int idx = i * 256 + tid;
        int r = idx / NEXP, e = idx % NEXP;
        float acc = pol_b[e];
        #pragma unroll
        for (int d = 0; d < HID; ++d) acc += zs[r][d] * pol_w[d * NEXP + e];
        logits[(row0 + r) * NEXP + e] = acc;
    }
}

// ---------------------------------------------------------------------------
// Kernel 2: TRIZ 40-expert GEMM via v_wmma_f32_16x16x32_f16.
// grid = (B/128, 40). Block = 256 threads = 8 waves; each wave owns a 16x64
// output tile. z tile (128x64 f16) and W_e^T (64x64 f16, [n][k]) staged in LDS.
// ---------------------------------------------------------------------------
__global__ void __launch_bounds__(256, 2) triz_wmma_kernel(
    const _Float16* __restrict__ zh, const float* __restrict__ triz_w,
    const float* __restrict__ triz_b, float* __restrict__ h)
{
    __shared__ _Float16 zsm[128 * HID];   // 16 KB, [row][k]
    __shared__ _Float16 wts[HID * HID];   //  8 KB, transposed: [n][k]
    __shared__ float    bs[HID];

    const int tid = threadIdx.x;
    const int e = blockIdx.y;
    const long row0 = (long)blockIdx.x * 128;

    // Stage z tile: 8192 halves, 4x v8h (b128) per thread
    {
        const v8h* zsrc = (const v8h*)(zh + row0 * HID);
        v8h* zdst = (v8h*)zsm;
        #pragma unroll
        for (int i = 0; i < 4; ++i) zdst[i * 256 + tid] = zsrc[i * 256 + tid];
    }
    // Stage W_e transposed (global [d][f] -> LDS [f][d]) + bias
    {
        const float* wsrc = triz_w + (long)e * HID * HID;
        #pragma unroll
        for (int i = 0; i < 16; ++i) {
            int idx = i * 256 + tid;
            int d = idx >> 6, f = idx & 63;
            wts[f * HID + d] = (_Float16)wsrc[idx];
        }
        if (tid < HID) bs[tid] = triz_b[e * HID + tid];
    }
    __syncthreads();

    const int wave  = tid >> 5;
    const int lane  = tid & 31;
    const int lhalf = lane >> 4;   // 0: lanes 0-15, 1: lanes 16-31
    const int l15   = lane & 15;
    const int mrow  = wave * 16;   // this wave's 16 rows

    // A fragments (16x32 f16 per ISA layout):
    //  lane<16 : M=lane,    halves = K{kc+0..7},  K{kc+16..23}
    //  lane>=16: M=lane-16, halves = K{kc+8..15}, K{kc+24..31}
    v16h a0, a1;
    {
        const _Float16* zr = zsm + (mrow + l15) * HID;
        const int kb = lhalf * 8;
        v8h lo0 = *(const v8h*)(zr + kb);
        v8h hi0 = *(const v8h*)(zr + kb + 16);
        v8h lo1 = *(const v8h*)(zr + 32 + kb);
        v8h hi1 = *(const v8h*)(zr + 32 + kb + 16);
        #pragma unroll
        for (int i = 0; i < 8; ++i) {
            a0[i] = lo0[i]; a0[i + 8] = hi0[i];
            a1[i] = lo1[i]; a1[i + 8] = hi1[i];
        }
    }

    #pragma unroll
    for (int nt = 0; nt < 4; ++nt) {
        const int n = nt * 16 + l15;
        // B fragments (32x16): lane column n = lane&15, K = lhalf*8 + {0..7}, lhalf*8 + 16 + {0..7}
        const _Float16* wr = wts + n * HID + lhalf * 8;
        v16h b0, b1;
        {
            v8h p0 = *(const v8h*)(wr);        // chunk0 K 8h+{0..7}, 16+8h+{0..7}
            v8h p1 = *(const v8h*)(wr + 16);
            v8h p2 = *(const v8h*)(wr + 32);   // chunk1 K 32+8h+{0..7}, 48+8h+{0..7}
            v8h p3 = *(const v8h*)(wr + 48);
            #pragma unroll
            for (int i = 0; i < 8; ++i) {
                b0[i] = p0[i]; b0[i + 8] = p1[i];
                b1[i] = p2[i]; b1[i + 8] = p3[i];
            }
        }
        // Accumulator seeded with bias (bias depends only on N = column)
        float bias = bs[n];
        v8f c;
        #pragma unroll
        for (int i = 0; i < 8; ++i) c[i] = bias;

        c = __builtin_amdgcn_wmma_f32_16x16x32_f16(false, a0, false, b0,
                                                   (short)0, c, false, false);
        c = __builtin_amdgcn_wmma_f32_16x16x32_f16(false, a1, false, b1,
                                                   (short)0, c, false, false);

        // ReLU + store. C/D layout: VGPR j -> (M = j + 8*lhalf, N = nt*16 + l15)
        const long brow = row0 + mrow + lhalf * 8;
        float* hp = h + (brow * NEXP + e) * HID + nt * 16 + l15;
        #pragma unroll
        for (int j = 0; j < 8; ++j) {
            float v = c[j] > 0.f ? c[j] : 0.f;
            hp[(long)j * (NEXP * HID)] = v;
        }
    }
}

// ---------------------------------------------------------------------------
// Kernel 3: softmax + Gumbel-argmax categorical (threefry2x32, key=[0,1]).
// Thread per row.
// ---------------------------------------------------------------------------
__device__ __forceinline__ unsigned rotl32(unsigned v, unsigned n) {
    return (v << n) | (v >> (32u - n));
}
__device__ __forceinline__ void threefry2x32(unsigned k0, unsigned k1,
                                             unsigned c0, unsigned c1,
                                             unsigned& o0, unsigned& o1)
{
    const unsigned ks[3] = {k0, k1, k0 ^ k1 ^ 0x1BD11BDAu};
    unsigned x0 = c0 + k0, x1 = c1 + k1;
    const unsigned R0[4] = {13, 15, 26, 6};
    const unsigned R1[4] = {17, 29, 16, 24};
    #pragma unroll
    for (int g = 0; g < 5; ++g) {
        const unsigned* R = (g & 1) ? R1 : R0;
        #pragma unroll
        for (int r = 0; r < 4; ++r) { x0 += x1; x1 = rotl32(x1, R[r]); x1 ^= x0; }
        x0 += ks[(g + 1) % 3];
        x1 += ks[(g + 2) % 3] + (unsigned)(g + 1);
    }
    o0 = x0; o1 = x1;
}

__global__ void __launch_bounds__(256) sample_kernel(
    const float* __restrict__ logits, float* __restrict__ probs,
    int* __restrict__ actions_i, float* __restrict__ actions_f)
{
    const long b = (long)blockIdx.x * 256 + threadIdx.x;
    const float* lg = logits + b * NEXP;

    float lv[NEXP];
    float mx = -3.4e38f;
    #pragma unroll
    for (int e = 0; e < NEXP; ++e) { lv[e] = lg[e]; mx = fmaxf(mx, lv[e]); }
    float sum = 0.f;
    float ex[NEXP];
    #pragma unroll
    for (int e = 0; e < NEXP; ++e) { ex[e] = __expf(lv[e] - mx); sum += ex[e]; }
    const float inv = 1.f / sum;

    float best = -3.4e38f;
    int bi = 0;
    #pragma unroll
    for (int e = 0; e < NEXP; ++e) {
        unsigned idx = (unsigned)(b * NEXP + e);
        unsigned r0, r1;
        threefry2x32(0u, 1u, idx, idx ^ 0x9E3779B9u, r0, r1);
        float u = __uint_as_float((r0 >> 9) | 0x3f800000u) - 1.0f;  // [0,1)
        float g = -logf(-logf(u + 1e-12f));
        float s = lv[e] + g;
        if (s > best) { best = s; bi = e; }
    }
    #pragma unroll
    for (int e = 0; e < NEXP; ++e) probs[b * NEXP + e] = ex[e] * inv;
    actions_i[b] = bi;
    actions_f[b] = (float)bi;
}

// ---------------------------------------------------------------------------
// Kernel 4: gather selected expert row of h, apply 64->5 decoder.
// ---------------------------------------------------------------------------
__global__ void __launch_bounds__(256) decode_kernel(
    const float* __restrict__ h, const int* __restrict__ actions,
    const float* __restrict__ dec_w, const float* __restrict__ dec_b,
    float* __restrict__ out)
{
    const long b = (long)blockIdx.x * 256 + threadIdx.x;
    const int a = actions[b];
    const float* hr = h + (b * NEXP + a) * HID;
    float acc[OUTD];
    #pragma unroll
    for (int o = 0; o < OUTD; ++o) acc[o] = dec_b[o];
    #pragma unroll
    for (int d = 0; d < HID; ++d) {
        float v = hr[d];
        #pragma unroll
        for (int o = 0; o < OUTD; ++o) acc[o] += v * dec_w[d * OUTD + o];
    }
    #pragma unroll
    for (int o = 0; o < OUTD; ++o) out[b * OUTD + o] = acc[o];
}

// ---------------------------------------------------------------------------
extern "C" void kernel_launch(void* const* d_in, const int* in_sizes, int n_in,
                              void* d_out, int out_size, void* d_ws, size_t ws_size,
                              hipStream_t stream) {
    const float* x      = (const float*)d_in[0];
    const float* enc_w  = (const float*)d_in[1];
    const float* enc_b  = (const float*)d_in[2];
    const float* triz_w = (const float*)d_in[3];
    const float* triz_b = (const float*)d_in[4];
    const float* pol_w  = (const float*)d_in[5];
    const float* pol_b  = (const float*)d_in[6];
    const float* dec_w  = (const float*)d_in[7];
    const float* dec_b  = (const float*)d_in[8];

    // d_out tuple layout: out[B,5] | probs[B,40] | actions[B] | h[B,40,64]
    float* out_p   = (float*)d_out;
    float* probs_p = out_p + (long)BATCH * OUTD;
    float* act_f_p = probs_p + (long)BATCH * NEXP;
    float* h_p     = act_f_p + (long)BATCH;

    // Workspace: zh (f16 B*64) | logits (f32 B*40) | actions_i (i32 B)
    char* ws = (char*)d_ws;
    _Float16* zh     = (_Float16*)ws;
    float*    logits = (float*)(ws + (size_t)BATCH * HID * sizeof(_Float16));
    int*      act_i  = (int*)(ws + (size_t)BATCH * HID * sizeof(_Float16)
                                 + (size_t)BATCH * NEXP * sizeof(float));

    enc_logits_kernel<<<BATCH / 64, 256, 0, stream>>>(
        x, enc_w, enc_b, pol_w, pol_b, zh, logits);

    dim3 tgrid(BATCH / 128, NEXP);
    triz_wmma_kernel<<<tgrid, 256, 0, stream>>>(zh, triz_w, triz_b, h_p);

    sample_kernel<<<BATCH / 256, 256, 0, stream>>>(logits, probs_p, act_i, act_f_p);

    decode_kernel<<<BATCH / 256, 256, 0, stream>>>(h_p, act_i, dec_w, dec_b, out_p);
}
